// PraxisRecurrent_49057116455201
// MI455X (gfx1250) — compile-verified, matrix operand order checked by reading, not run on hardware
//
#include <hip/hip_runtime.h>
#include <hip/hip_bf16.h>
#include <cstdint>
#include <cstddef>

// ---------------------------------------------------------------------------
// Types for CDNA5 WMMA (wave32, v_wmma_f32_16x16x32_bf16)
// ---------------------------------------------------------------------------
typedef __bf16 bf16_t;
typedef __attribute__((ext_vector_type(16))) __bf16 v16bf;
typedef __attribute__((ext_vector_type(8)))  float  v8f;
typedef __attribute__((ext_vector_type(4)))  __bf16 v4bf;

constexpr int CB  = 4;        // batch
constexpr int CS  = 2048;     // sequence
constexpr int CD  = 1024;     // model dim
constexpr int CE  = 8;        // experts
constexpr int CK2 = 2 * CD;   // 2048 (hidden+gate)
constexpr int NCH = 8;        // scan chunks along S
constexpr int CHL = CS / NCH; // 256

// ---------------------------------------------------------------------------
// RMSNorm: one block per (b,s) row; writes x in bf16 for the WMMA GEMMs.
// ---------------------------------------------------------------------------
__global__ __launch_bounds__(256)
void rmsnorm_kernel(const float* __restrict__ in, const float* __restrict__ nw,
                    bf16_t* __restrict__ xout)
{
    const int row = blockIdx.x;               // b*S + s
    const int tid = threadIdx.x;
    float4 v = ((const float4*)(in + (size_t)row * CD))[tid];
    float ss = v.x*v.x + v.y*v.y + v.z*v.z + v.w*v.w;
    #pragma unroll
    for (int o = 16; o > 0; o >>= 1) ss += __shfl_xor(ss, o, 32);
    __shared__ float red[8];
    if ((tid & 31) == 0) red[tid >> 5] = ss;
    __syncthreads();
    float tot = 0.0f;
    #pragma unroll
    for (int i = 0; i < 8; ++i) tot += red[i];
    const float scale = rsqrtf(tot * (1.0f / CD) + 1e-6f);
    const float4 w = ((const float4*)nw)[tid];
    v4bf o4;
    o4[0] = (bf16_t)(v.x * scale * w.x);
    o4[1] = (bf16_t)(v.y * scale * w.y);
    o4[2] = (bf16_t)(v.z * scale * w.z);
    o4[3] = (bf16_t)(v.w * scale * w.w);
    ((v4bf*)(xout + (size_t)row * CD))[tid] = o4;
}

__global__ void init_kernel(float* __restrict__ macc)
{
    const int i = blockIdx.x * 256 + threadIdx.x;
    if (i < CB * CD) macc[i] = 0.0f;
}

// Column sums of x over S (for the router's mean pooling), chunked + atomic.
__global__ __launch_bounds__(256)
void colsum_kernel(const bf16_t* __restrict__ x, float* __restrict__ macc)
{
    const int b = blockIdx.z, ch = blockIdx.y;
    const int d = blockIdx.x * 256 + threadIdx.x;
    const bf16_t* p = x + ((size_t)b * CS + (size_t)ch * CHL) * CD + d;
    float s = 0.0f;
    for (int i = 0; i < CHL; ++i) s += (float)p[(size_t)i * CD];
    atomicAdd(&macc[b * CD + d], s);
}

// Router: 32 waves, one per (b,e) dot product; softmax per batch; aux=0.
__global__ __launch_bounds__(1024)
void router_kernel(const float* __restrict__ macc, const float* __restrict__ rw,
                   const float* __restrict__ rb, float* __restrict__ probs,
                   float* __restrict__ auxp)
{
    const int wave = threadIdx.x >> 5, lane = threadIdx.x & 31;
    const int b = wave >> 3, e = wave & 7;
    const float* mp = macc + b * CD;
    const float* wp = rw + e * CD;
    float s = 0.0f;
    for (int d = lane; d < CD; d += 32) s += mp[d] * wp[d];
    #pragma unroll
    for (int o = 16; o > 0; o >>= 1) s += __shfl_xor(s, o, 32);
    __shared__ float lg[CB][CE];
    if (lane == 0) lg[b][e] = s * (1.0f / CS) + rb[e];
    __syncthreads();
    if (threadIdx.x < CB) {
        const int bb = threadIdx.x;
        float mx = lg[bb][0];
        #pragma unroll
        for (int i = 1; i < CE; ++i) mx = fmaxf(mx, lg[bb][i]);
        float ex[CE], den = 0.0f;
        #pragma unroll
        for (int i = 0; i < CE; ++i) { ex[i] = __expf(lg[bb][i] - mx); den += ex[i]; }
        const float inv = 1.0f / den;
        #pragma unroll
        for (int i = 0; i < CE; ++i) probs[bb * CE + i] = ex[i] * inv;
    }
    if (threadIdx.x == 0) *auxp = 0.0f;
}

// Mix expert weights: dst[b] = sum_e probs[b,e] * wsrc[e]; read experts once.
__global__ __launch_bounds__(256)
void mix_kernel(const float* __restrict__ wsrc, const float* __restrict__ probs,
                bf16_t* __restrict__ dst, int R)
{
    const size_t n = (size_t)R * CD;
    const size_t idx = (size_t)blockIdx.x * 256 + threadIdx.x;
    if (idx >= n) return;
    float vals[CE];
    #pragma unroll
    for (int e = 0; e < CE; ++e) vals[e] = wsrc[(size_t)e * n + idx];
    #pragma unroll
    for (int b = 0; b < CB; ++b) {
        float s = 0.0f;
        #pragma unroll
        for (int e = 0; e < CE; ++e) s += probs[b * CE + e] * vals[e];
        dst[(size_t)b * n + idx] = (bf16_t)s;
    }
}

// ---------------------------------------------------------------------------
// WMMA GEMM (NT): C[b] = A[b] (MxK, row-major) * B[b]^T (B is NxK row-major).
// Per-lane fragments are contiguous 32B loads in a contiguous k order shared by A and B (not AMD's table
// order): lane L reads row (L%16), K-offset (L/16)*16, 16 bf16 elements.
// Each wave computes a 64x64 tile (4x4 WMMA tiles); 8 waves -> 128x256 block.
// ---------------------------------------------------------------------------
template<bool ADD_INPUT>
__global__ __launch_bounds__(256)
void wmma_gemm_nt(const bf16_t* __restrict__ Amat,
                  const bf16_t* __restrict__ Bmat,
                  const float*  __restrict__ addsrc,
                  float*  __restrict__ outF,
                  bf16_t* __restrict__ outB,
                  int M, int N, int K)
{
    const int wave = threadIdx.x >> 5;
    const int lane = threadIdx.x & 31;
    const int mt = blockIdx.y * 2 + (wave >> 2);
    const int nt = blockIdx.x * 4 + (wave & 3);
    const int b  = blockIdx.z;
    const int m0 = mt * 64, n0 = nt * 64;
    const int lr = lane & 15;
    const int kh = (lane >> 4) << 4;   // 0 or 16

    const bf16_t* Ab = Amat + (size_t)b * M * K;
    const bf16_t* Bb = Bmat + (size_t)b * N * K;
    const bf16_t* ap = Ab + (size_t)(m0 + lr) * K + kh;
    const bf16_t* bp = Bb + (size_t)(n0 + lr) * K + kh;

    v8f acc[4][4];
    #pragma unroll
    for (int i = 0; i < 4; ++i)
        #pragma unroll
        for (int j = 0; j < 4; ++j)
            #pragma unroll
            for (int t = 0; t < 8; ++t) acc[i][j][t] = 0.0f;

    for (int k0 = 0; k0 < K; k0 += 32) {
        v16bf afr[4], bfr[4];
        #pragma unroll
        for (int i = 0; i < 4; ++i)
            afr[i] = *(const v16bf*)(ap + (size_t)(i * 16) * K + k0);
        #pragma unroll
        for (int j = 0; j < 4; ++j)
            bfr[j] = *(const v16bf*)(bp + (size_t)(j * 16) * K + k0);
        if (k0 + 256 < K) {
            // locality=3 -> nearest-scope global_prefetch_b8 (WGP: pull into
            // all cache levels), not the SCOPE_SYS variant locality=0 gives.
            __builtin_prefetch(ap + k0 + 256, 0, 3);
            __builtin_prefetch(bp + k0 + 256, 0, 3);
        }
        #pragma unroll
        for (int i = 0; i < 4; ++i)
            #pragma unroll
            for (int j = 0; j < 4; ++j)
                acc[i][j] = __builtin_amdgcn_wmma_f32_16x16x32_bf16(
                    false, afr[i], false, bfr[j], (short)0, acc[i][j],
                    false, false);
    }

    // C/D layout: vgpr v, lane L -> row = 8*(L/16)+v, col = L%16.
    const int rbase = m0 + ((lane >> 4) << 3);
    const int cbase = n0 + (lane & 15);
    #pragma unroll
    for (int i = 0; i < 4; ++i) {
        #pragma unroll
        for (int vv = 0; vv < 8; ++vv) {
            const int r = rbase + i * 16 + vv;
            const size_t rowoff = ((size_t)b * M + r) * N;
            #pragma unroll
            for (int j = 0; j < 4; ++j) {
                const int c = cbase + j * 16;
                const float val = acc[i][j][vv];
                if (ADD_INPUT) outF[rowoff + c] = val + addsrc[rowoff + c];
                else           outB[rowoff + c] = (bf16_t)val;
            }
        }
    }
}

// ---------------------------------------------------------------------------
// Recurrence: h_t = sigmoid(-gate)*h_{t-1} + sigmoid(gate)*g~(hidden)
// (exactly exp() of the reference's log-space scan). Chunked 3-phase scan.
// ---------------------------------------------------------------------------
__device__ __forceinline__ void coef_val(float hid, float gt, float& c, float& v)
{
    const float sg = 1.0f / (1.0f + __expf(-gt));           // sigmoid(gate)
    c = 1.0f - sg;                                          // exp(-softplus(gate))
    const float gf = (hid >= 0.0f) ? (hid + 0.5f)
                                   : (1.0f / (1.0f + __expf(-hid)));
    v = sg * gf;                                            // exp(-softplus(-g)+log_g(h))
}

__global__ __launch_bounds__(256)
void scan_part1(const bf16_t* __restrict__ hg, float* __restrict__ cA,
                float* __restrict__ cBv)
{
    const int b = blockIdx.z, ch = blockIdx.y;
    const int d = blockIdx.x * 256 + threadIdx.x;
    const bf16_t* base = hg + ((size_t)b * CS + (size_t)ch * CHL) * CK2;
    float A = 1.0f, Bv = 0.0f;
    for (int i = 0; i < CHL; ++i) {
        const bf16_t* row = base + (size_t)i * CK2;
        float c, v;
        coef_val((float)row[d], (float)row[CD + d], c, v);
        A *= c;  Bv = c * Bv + v;
    }
    const size_t o = ((size_t)b * NCH + ch) * CD + d;
    cA[o] = A;  cBv[o] = Bv;
}

__global__ __launch_bounds__(256)
void scan_part2(const float* __restrict__ cA, const float* __restrict__ cBv,
                float* __restrict__ hstart)
{
    const int gid = blockIdx.x * 256 + threadIdx.x;  // over CB*CD
    const int b = gid >> 10, d = gid & (CD - 1);
    float h = 0.0f;
    #pragma unroll
    for (int ch = 0; ch < NCH; ++ch) {
        const size_t o = ((size_t)b * NCH + ch) * CD + d;
        hstart[o] = h;
        h = cA[o] * h + cBv[o];
    }
}

__global__ __launch_bounds__(256)
void scan_part3(const bf16_t* __restrict__ hg, const float* __restrict__ hstart,
                bf16_t* __restrict__ hout, float* __restrict__ newstate)
{
    const int b = blockIdx.z, ch = blockIdx.y;
    const int d = blockIdx.x * 256 + threadIdx.x;
    const bf16_t* base = hg + ((size_t)b * CS + (size_t)ch * CHL) * CK2;
    bf16_t* hb = hout + ((size_t)b * CS + (size_t)ch * CHL) * CD;
    float h = hstart[((size_t)b * NCH + ch) * CD + d];
    for (int i = 0; i < CHL; ++i) {
        const bf16_t* row = base + (size_t)i * CK2;
        float c, v;
        coef_val((float)row[d], (float)row[CD + d], c, v);
        h = c * h + v;
        hb[(size_t)i * CD + d] = (bf16_t)h;
    }
    if (ch == NCH - 1) newstate[(size_t)b * CD + d] = h;
}

// ---------------------------------------------------------------------------
// Host launcher
// ---------------------------------------------------------------------------
extern "C" void kernel_launch(void* const* d_in, const int* in_sizes, int n_in,
                              void* d_out, int out_size, void* d_ws, size_t ws_size,
                              hipStream_t stream)
{
    (void)in_sizes; (void)n_in; (void)out_size; (void)ws_size;

    const float* inputs = (const float*)d_in[0];
    const float* norm_w = (const float*)d_in[1];
    const float* rw     = (const float*)d_in[2];
    const float* rb     = (const float*)d_in[3];
    const float* w_hg   = (const float*)d_in[4];
    const float* w_out  = (const float*)d_in[5];

    float* out      = (float*)d_out;
    float* newstate = out + (size_t)CB * CS * CD;
    float* auxp     = newstate + (size_t)CB * CD;

    char* ws = (char*)d_ws;
    size_t off = 0;
    auto take = [&](size_t bytes) -> char* {
        char* p = ws + off;
        off = (off + bytes + 255) & ~(size_t)255;
        return p;
    };
    bf16_t* x_bf    = (bf16_t*)take((size_t)CB * CS * CD * 2);   // 16.8 MB
    bf16_t* hg_bf   = (bf16_t*)take((size_t)CB * CS * CK2 * 2);  // 33.6 MB
    bf16_t* h_bf    = (bf16_t*)take((size_t)CB * CS * CD * 2);   // 16.8 MB
    bf16_t* whg_bf  = (bf16_t*)take((size_t)CB * CK2 * CD * 2);  // 16.8 MB
    bf16_t* wout_bf = (bf16_t*)take((size_t)CB * CD * CD * 2);   //  8.4 MB
    float*  macc    = (float*)take((size_t)CB * CD * 4);
    float*  probs   = (float*)take((size_t)CB * CE * 4);
    float*  cA      = (float*)take((size_t)CB * NCH * CD * 4);
    float*  cBv     = (float*)take((size_t)CB * NCH * CD * 4);
    float*  hstart  = (float*)take((size_t)CB * NCH * CD * 4);

    init_kernel<<<(CB * CD + 255) / 256, 256, 0, stream>>>(macc);
    rmsnorm_kernel<<<CB * CS, 256, 0, stream>>>(inputs, norm_w, x_bf);
    colsum_kernel<<<dim3(CD / 256, NCH, CB), 256, 0, stream>>>(x_bf, macc);
    router_kernel<<<1, 1024, 0, stream>>>(macc, rw, rb, probs, auxp);
    mix_kernel<<<(int)(((size_t)CK2 * CD + 255) / 256), 256, 0, stream>>>(
        w_hg, probs, whg_bf, CK2);
    mix_kernel<<<(int)(((size_t)CD * CD + 255) / 256), 256, 0, stream>>>(
        w_out, probs, wout_bf, CD);
    // GEMM1: hg = x @ W_hg^T  (M=2048, N=2048, K=1024) per batch
    wmma_gemm_nt<false><<<dim3(CK2 / 256, CS / 128, CB), 256, 0, stream>>>(
        x_bf, whg_bf, nullptr, nullptr, hg_bf, CS, CK2, CD);
    // Parallel recurrence along S
    scan_part1<<<dim3(CD / 256, NCH, CB), 256, 0, stream>>>(hg_bf, cA, cBv);
    scan_part2<<<(CB * CD + 255) / 256, 256, 0, stream>>>(cA, cBv, hstart);
    scan_part3<<<dim3(CD / 256, NCH, CB), 256, 0, stream>>>(hg_bf, hstart, h_bf,
                                                            newstate);
    // GEMM2: out = h @ W_out^T + inputs  (M=2048, N=1024, K=1024) per batch
    wmma_gemm_nt<true><<<dim3(CD / 256, CS / 128, CB), 256, 0, stream>>>(
        h_bf, wout_bf, inputs, out, nullptr, CS, CD, CD);
}
